// LocalAttentionND_23304492548289
// MI455X (gfx1250) — matrix operand order repaired and checked by reading, not compiled, not run on hardware
//
#include <hip/hip_runtime.h>
#include <hip/hip_fp16.h>
#include <hip/hip_bf16.h>

// ---------------------------------------------------------------------------
// Problem constants (from reference: B=2, 64x64 grid, C=256, H=4, D=64, 7x7)
// ---------------------------------------------------------------------------
static constexpr int NTOK   = 2 * 64 * 64;   // 8192 tokens
static constexpr int CCH    = 256;           // channels
static constexpr int NH     = 4;             // heads
static constexpr int HD     = 64;            // head dim
static constexpr int QKVW   = 3 * CCH;       // 768
static constexpr float EPSF = 1e-6f;
static constexpr float MAXDIST = 4.242640687119285f; // sqrt(18)

typedef __attribute__((ext_vector_type(16))) _Float16 v16h;
typedef __attribute__((ext_vector_type(8)))  float    v8f;

__device__ __forceinline__ float sigf(float z) { return 1.0f / (1.0f + __expf(-z)); }
__device__ __forceinline__ float siluf(float z) { return z * sigf(z); }

// ---------------------------------------------------------------------------
// Block-tiled WMMA GEMM: Out[M x N] = act(A[M x K] @ B[K x N] (+ bias))
//
//   Block = 256 threads (8 waves), block tile = 128 (M) x 64 (N).
//   Wave w owns rows m0 = blkM*128 + w*16; 4 accumulators cover 64 columns.
//   All leading dimensions are template constants so every VMEM access is
//   base-register + 24-bit immediate (no per-element 64-bit address math).
//
//   Per K-step (32): the 32x64 B slab is staged into LDS pre-swizzled into
//   WMMA B-fragment order. Each thread loads 2 x float4 (global_load_b128,
//   coalesced along N), packs to f16, and scatters 8 b16 LDS stores; each
//   wave then fetches a fragment with 2 x ds_load_b128. A fragments are
//   per-wave float4 loads from global, reused across the 4 N-tiles.
//
// Fragment layouts used here (f16, 16x16x32; same contiguous k order for A and B):
//   A 16x32: lane%16 -> M row; half i = K[hi*16 + i]
//   B 32x16: lane%16 -> N col, hi -> K half; half i = row K[hi*16 + i]
//   C/D 16x16 f32: vgpr j -> M = j + hi*8, N = lane%16
// MODE 0: silu(z);  MODE 1: sigmoid(silu(z + bias))
// ---------------------------------------------------------------------------
template <int K, int LDA, int LDB, int LDO, int MODE>
__global__ __launch_bounds__(256)
void gemm_act_wmma(const float* __restrict__ A,
                   const float* __restrict__ B,
                   const float* __restrict__ bias,
                   float* __restrict__ Out) {
    __shared__ _Float16 Bs[4 * 32 * 16];   // [ntile][lane][half] = 4 KB

    const int tid  = threadIdx.x;
    const int wave = tid >> 5;
    const int lane = tid & 31;
    const int lrow = lane & 15;
    const int hi   = lane >> 4;

    const int m0 = blockIdx.y * 128 + wave * 16;
    const int n0 = blockIdx.x * 64;

    // A fragment base: row (m0+lrow), K offset hi*16 folded in.
    const float* Arow = A + (size_t)(m0 + lrow) * LDA + hi * 16;

    // B staging: 32x64 slab = 512 float4 chunks; thread owns chunks
    // {tid, tid+256}: kk = tid>>4 (+16 for the 2nd), nn = (tid&15)*4.
    const int kk0 = tid >> 4;            // 0..15  (half index i = kk0)
    const int nn  = (tid & 15) * 4;      // 0..60
    const int t_s = nn >> 4;             // N-tile of this chunk
    const int ls0 = nn & 15;             // first lane_s (kk<16 half)
    const float* Bp = B + (size_t)kk0 * LDB + n0 + nn;
    _Float16* s0 = &Bs[(t_s * 32 + ls0) * 16 + kk0];  // kk<16 rows
    _Float16* s1 = s0 + 16 * 16;                      // kk>=16 rows (lane_s+16)

    v8f acc[4] = {};

#pragma unroll
    for (int k0 = 0; k0 < K; k0 += 32) {
        // global loads for this step (issue before barrier; overlap prev WMMAs)
        const float4 f0 = *(const float4*)(Bp + (size_t)k0 * LDB);
        const float4 f1 = *(const float4*)(Bp + (size_t)(k0 + 16) * LDB);
        const float4 a0 = *(const float4*)(Arow + k0);
        const float4 a1 = *(const float4*)(Arow + k0 + 4);
        const float4 a2 = *(const float4*)(Arow + k0 + 8);
        const float4 a3 = *(const float4*)(Arow + k0 + 12);
        if (k0 + 32 < K) __builtin_prefetch(Arow + k0 + 32, 0, 0);

        __syncthreads();   // previous step's fragment reads complete
        // scatter into fragment order (consecutive lane_s, fixed half kk0)
        s0[0 * 16] = (_Float16)f0.x; s0[1 * 16] = (_Float16)f0.y;
        s0[2 * 16] = (_Float16)f0.z; s0[3 * 16] = (_Float16)f0.w;
        s1[0 * 16] = (_Float16)f1.x; s1[1 * 16] = (_Float16)f1.y;
        s1[2 * 16] = (_Float16)f1.z; s1[3 * 16] = (_Float16)f1.w;
        __syncthreads();   // staging visible to all waves

        v16h a;
        a[0]  = (_Float16)a0.x; a[1]  = (_Float16)a0.y;
        a[2]  = (_Float16)a0.z; a[3]  = (_Float16)a0.w;
        a[4]  = (_Float16)a1.x; a[5]  = (_Float16)a1.y;
        a[6]  = (_Float16)a1.z; a[7]  = (_Float16)a1.w;
        a[8]  = (_Float16)a2.x; a[9]  = (_Float16)a2.y;
        a[10] = (_Float16)a2.z; a[11] = (_Float16)a2.w;
        a[12] = (_Float16)a3.x; a[13] = (_Float16)a3.y;
        a[14] = (_Float16)a3.z; a[15] = (_Float16)a3.w;

#pragma unroll
        for (int t = 0; t < 4; ++t) {
            v16h b;
            const _Float16* bp = &Bs[(t * 32 + lane) * 16]; // 32B contiguous
#pragma unroll
            for (int i = 0; i < 16; ++i) b[i] = bp[i];
            acc[t] = __builtin_amdgcn_wmma_f32_16x16x32_f16(
                /*neg_a=*/false, a, /*neg_b=*/false, b,
                /*c_mod=*/(short)0, acc[t], /*reuse_a=*/false, /*reuse_b=*/false);
        }
    }

    // ---- epilogue: activation + store ----------------------------------
#pragma unroll
    for (int t = 0; t < 4; ++t) {
        const int col = n0 + t * 16 + lrow;
        const float bval = bias ? bias[col] : 0.0f;
#pragma unroll
        for (int j = 0; j < 8; ++j) {
            float z = acc[t][j] + bval;
            float s = siluf(z);
            if (MODE == 1) s = sigf(s);
            Out[(size_t)(m0 + j + hi * 8) * LDO + col] = s;
        }
    }
}

// ---------------------------------------------------------------------------
// wp = silu(x @ W_wp + b_wp); width = sig(wp[:4])*maxdist+0.5,
// sharp = sig(wp[4:])*9.5+0.5.  One thread per (token, output). 8 outputs.
// ---------------------------------------------------------------------------
__global__ __launch_bounds__(256)
void wp_kernel(const float* __restrict__ x, const float* __restrict__ Wwp,
               const float* __restrict__ bwp,
               float* __restrict__ widthb, float* __restrict__ sharpb) {
    int tid = blockIdx.x * blockDim.x + threadIdx.x;
    int tok = tid >> 3;
    int o   = tid & 7;
    if (tok >= NTOK) return;
    const float* xr = x + (size_t)tok * CCH;
    float acc = bwp[o];
    for (int k = 0; k < CCH; ++k) acc += xr[k] * Wwp[k * 8 + o];
    float sg = sigf(siluf(acc));
    if (o < 4) widthb[tok * 4 + o]       = sg * MAXDIST + 0.5f;
    else       sharpb[tok * 4 + (o - 4)] = sg * 9.5f + 0.5f;
}

// ---------------------------------------------------------------------------
// In-place RMSNorm(D=64) + RoPE on q and k slices of qkv.
// One wave per (token, head); lane owns dims {lane, lane+32}.
// NOTE: reference applies RoPE with pos = head index (H axis), half = 32.
// ---------------------------------------------------------------------------
__global__ __launch_bounds__(256)
void qk_norm_rope(float* __restrict__ qkv,
                  const float* __restrict__ wq, const float* __restrict__ wk) {
    int wid  = (blockIdx.x * blockDim.x + threadIdx.x) >> 5;
    int lane = threadIdx.x & 31;
    int tok  = wid >> 2;
    int h    = wid & 3;
    if (tok >= NTOK) return;

    const float k_ln = logf(10000.0f) / 32.0f;
    float freq = __expf(-(float)lane * k_ln);
    float ang  = (float)h * freq;
    float cs   = __cosf(ang);
    float sn   = __sinf(ang);

    for (int which = 0; which < 2; ++which) {
        float* base = qkv + (size_t)tok * QKVW + which * CCH + h * HD;
        const float* w = (which == 0) ? wq : wk;
        float x1 = base[lane], x2 = base[lane + 32];
        float ss = x1 * x1 + x2 * x2;
#pragma unroll
        for (int m = 16; m >= 1; m >>= 1) ss += __shfl_xor(ss, m, 32);
        float rinv = rsqrtf(ss * (1.0f / 64.0f) + EPSF);
        x1 = x1 * rinv * w[lane];
        x2 = x2 * rinv * w[lane + 32];
        base[lane]      = x1 * cs - x2 * sn;
        base[lane + 32] = x1 * sn + x2 * cs;
    }
}

// ---------------------------------------------------------------------------
// 7x7 windowed attention with distance soft-mask, online softmax.
// One wave per (token, head); lane owns dims {lane, lane+32}.
// Score reduction across the wave via shfl_xor (wave32).
// ---------------------------------------------------------------------------
__global__ __launch_bounds__(256)
void attn_kernel(const float* __restrict__ qkv,
                 const float* __restrict__ widthb, const float* __restrict__ sharpb,
                 float* __restrict__ attn_out) {
    int wid  = (blockIdx.x * blockDim.x + threadIdx.x) >> 5;
    int lane = threadIdx.x & 31;
    int tok  = wid >> 2;
    int h    = wid & 3;
    if (tok >= NTOK) return;
    int b   = tok >> 12;       // / 4096
    int rem = tok & 4095;
    int y   = rem >> 6;
    int xg  = rem & 63;

    const float scale = 0.125f; // 64^-0.5
    const float* qp = qkv + (size_t)tok * QKVW + h * HD;
    float q0 = qp[lane], q1 = qp[lane + 32];
    float width = widthb[tok * 4 + h];
    float sharp = sharpb[tok * 4 + h];

    float m = -1e30f, ssum = 0.0f, a0 = 0.0f, a1 = 0.0f;
#pragma unroll 1
    for (int w = 0; w < 49; ++w) {
        int di = w / 7 - 3, dj = w % 7 - 3;
        int yy = y + di, xx = xg + dj;
        bool ok = (yy >= 0) && (yy < 64) && (xx >= 0) && (xx < 64);
        float k0 = 0.f, k1 = 0.f, v0 = 0.f, v1 = 0.f;
        if (ok) {
            int t2 = (b << 12) + (yy << 6) + xx;
            const float* kp = qkv + (size_t)t2 * QKVW + CCH + h * HD;
            const float* vp = qkv + (size_t)t2 * QKVW + 2 * CCH + h * HD;
            k0 = kp[lane]; k1 = kp[lane + 32];
            v0 = vp[lane]; v1 = vp[lane + 32];
        }
        float part = q0 * k0 + q1 * k1;
#pragma unroll
        for (int sh = 16; sh >= 1; sh >>= 1) part += __shfl_xor(part, sh, 32);
        float score = part * scale;
        float rd  = sqrtf((float)(di * di + dj * dj));
        float msk = sigf((width - rd) * sharp);
        score -= (1.0f - msk) * 10000.0f;
        if (score > m) {
            float corr = __expf(m - score);
            a0 *= corr; a1 *= corr; ssum *= corr;
            m = score;
        }
        float p = __expf(score - m);
        ssum += p;
        a0 += p * v0; a1 += p * v1;
    }
    float inv = 1.0f / ssum;
    float* op = attn_out + (size_t)tok * CCH + h * HD;
    op[lane]      = a0 * inv;
    op[lane + 32] = a1 * inv;
}

// ---------------------------------------------------------------------------
// RMSNorm(C=256) of attention output + gated merge:
//   merged = gate * v + (1 - gate) * rmsnorm(out, w_onorm)
// Block of 256 threads per token; LDS tree reduction. merged overwrites gate.
// ---------------------------------------------------------------------------
__global__ __launch_bounds__(256)
void merge_kernel(const float* __restrict__ attn, float* __restrict__ gatebuf,
                  const float* __restrict__ qkv, const float* __restrict__ wo) {
    __shared__ float red[256];
    int tok = blockIdx.x;
    int c   = threadIdx.x;
    float o = attn[(size_t)tok * CCH + c];
    red[c] = o * o;
    __syncthreads();
    for (int s = 128; s >= 1; s >>= 1) {
        if (c < s) red[c] += red[c + s];
        __syncthreads();
    }
    float rinv  = rsqrtf(red[0] * (1.0f / 256.0f) + EPSF);
    float onorm = o * rinv * wo[c];
    float g = gatebuf[(size_t)tok * CCH + c];
    float v = qkv[(size_t)tok * QKVW + 2 * CCH + c];
    gatebuf[(size_t)tok * CCH + c] = g * v + (1.0f - g) * onorm;
}

// ---------------------------------------------------------------------------
// Launch
// ---------------------------------------------------------------------------
extern "C" void kernel_launch(void* const* d_in, const int* in_sizes, int n_in,
                              void* d_out, int out_size, void* d_ws, size_t ws_size,
                              hipStream_t stream) {
    const float* x      = (const float*)d_in[0];  // (8192, 256)
    const float* W_qkv  = (const float*)d_in[1];  // (256, 768)
    const float* w_qn   = (const float*)d_in[2];  // (64,)
    const float* w_kn   = (const float*)d_in[3];  // (64,)
    const float* W_wp   = (const float*)d_in[4];  // (256, 8)
    const float* b_wp   = (const float*)d_in[5];  // (8,)
    const float* w_on   = (const float*)d_in[6];  // (256,)
    const float* W_out  = (const float*)d_in[7];  // (256, 256)
    const float* W_gate = (const float*)d_in[8];  // (256, 256)
    const float* b_gate = (const float*)d_in[9];  // (256,)
    float* out = (float*)d_out;                   // (8192, 256)

    float* ws      = (float*)d_ws;
    float* qkv     = ws;                                // 8192*768
    float* attn    = qkv + (size_t)NTOK * QKVW;         // 8192*256
    float* gatebuf = attn + (size_t)NTOK * CCH;         // 8192*256 (then merged)
    float* widthb  = gatebuf + (size_t)NTOK * CCH;      // 8192*4
    float* sharpb  = widthb + (size_t)NTOK * NH;        // 8192*4

    // 1) qkv = silu(x @ W_qkv)    [WMMA, 128x64 block tile]
    dim3 gqkv(QKVW / 64, NTOK / 128);
    gemm_act_wmma<CCH, CCH, QKVW, QKVW, 0>
        <<<gqkv, 256, 0, stream>>>(x, W_qkv, nullptr, qkv);
    // 2) width / sharpness projection
    wp_kernel<<<(NTOK * 8) / 256, 256, 0, stream>>>(x, W_wp, b_wp, widthb, sharpb);
    // 3) in-place RMSNorm + RoPE on q, k    (wave per token-head)
    qk_norm_rope<<<(NTOK * NH * 32) / 256, 256, 0, stream>>>(qkv, w_qn, w_kn);
    // 4) windowed attention with soft mask  (wave per token-head)
    attn_kernel<<<(NTOK * NH * 32) / 256, 256, 0, stream>>>(qkv, widthb, sharpb, attn);
    // 5) gate = sigmoid(silu(v @ W_gate + b_gate))    [WMMA]
    dim3 gsq(CCH / 64, NTOK / 128);
    gemm_act_wmma<CCH, QKVW, CCH, CCH, 1>
        <<<gsq, 256, 0, stream>>>(qkv + 2 * CCH, W_gate, b_gate, gatebuf);
    // 6) merged = gate*v + (1-gate)*rmsnorm(attn)
    merge_kernel<<<NTOK, 256, 0, stream>>>(attn, gatebuf, qkv, w_on);
    // 7) out = silu(merged @ W_out)    [WMMA]
    gemm_act_wmma<CCH, CCH, CCH, CCH, 0>
        <<<gsq, 256, 0, stream>>>(gatebuf, W_out, nullptr, out);
}
